// SwitchHeadCore_17205638988280
// MI455X (gfx1250) — matrix operand order repaired and checked by reading, not compiled, not run on hardware
//
#include <hip/hip_runtime.h>
#include <hip/hip_bf16.h>
#include <math.h>

#define B_ 4
#define S_ 2048
#define DM 1024
#define H_ 8
#define DH 128
#define E_ 8
#define RE 7
#define TOK (B_*S_)   /* 8192 tokens */
#define HE (H_*E_)    /* 64 expert matrices */

typedef __bf16 bf16;
typedef __attribute__((ext_vector_type(16))) __bf16 v16bf;
typedef __attribute__((ext_vector_type(8)))  __bf16 v8bf;
typedef __attribute__((ext_vector_type(8)))  float  v8f;
typedef __attribute__((ext_vector_type(4)))  int    v4i;

// ---- gfx1250 async global->LDS copy (ASYNCcnt) with safe fallback ----------
#if defined(__has_builtin)
#if __has_builtin(__builtin_amdgcn_global_load_async_to_lds_b128) && \
    __has_builtin(__builtin_amdgcn_s_wait_asynccnt)
#define USE_ASYNC_LDS 1
#endif
#endif
#ifndef USE_ASYNC_LDS
#define USE_ASYNC_LDS 0
#endif

static __device__ __forceinline__ void stage16(bf16* lds_dst, const bf16* gsrc) {
#if USE_ASYNC_LDS
  __builtin_amdgcn_global_load_async_to_lds_b128(
      (__attribute__((address_space(1))) v4i*)gsrc,
      (__attribute__((address_space(3))) v4i*)lds_dst, 0, 0);
#else
  *(v8bf*)lds_dst = *(const v8bf*)gsrc;
#endif
}
static __device__ __forceinline__ void stage_wait() {
#if USE_ASYNC_LDS
  __builtin_amdgcn_s_wait_asynccnt(0);
#endif
}

static __device__ __forceinline__ v8f wmma_bf16(v16bf a, v16bf b, v8f c) {
  // D = A(16x32 bf16) * B(32x16 bf16) + C(16x16 f32)
  return __builtin_amdgcn_wmma_f32_16x16x32_bf16(false, a, false, b, (short)0, c, false, false);
}

// A-fragment (16x32): lane L holds row m=L%16; kh=L/16 selects K halves
// a[i]   <-> K = kk + kh*16 + i       (i<8)
// a[8+i] <-> K = kk + kh*16 + 8 + i
static __device__ __forceinline__ v16bf load_a_f32(const float* row, int kk, int kh) {
  const float* p0 = row + kk + kh*16;
  const float* p1 = p0 + 8;
  v16bf a;
#pragma unroll
  for (int i = 0; i < 8; ++i) { a[i] = (bf16)p0[i]; a[8+i] = (bf16)p1[i]; }
  return a;
}

static __device__ __forceinline__ v16bf load_a_bf16(const bf16* row, int kk, int kh) {
  v8bf lo = *(const v8bf*)(row + kk + kh*16);
  v8bf hi = *(const v8bf*)(row + kk + kh*16 + 8);
  v16bf a;
#pragma unroll
  for (int i = 0; i < 8; ++i) { a[i] = lo[i]; a[8+i] = hi[i]; }
  return a;
}

// ---------------------------------------------------------------- init ------
__global__ void k_init(int* cnt_v, int* cnt_o) {
  int t = threadIdx.x;
  if (t < H_*RE) { cnt_v[t] = 0; cnt_o[t] = 0; }
}

// -------------------------------------------- weight transpose to bf16 ------
// src [HE][R][C] f32 -> dst [HE][C][R] bf16
__global__ void k_transpose(const float* src, bf16* dst, int R, int C) {
  size_t i = (size_t)blockIdx.x * blockDim.x + threadIdx.x;
  size_t total = (size_t)HE * R * C;
  if (i >= total) return;
  int c = (int)(i % C);
  size_t t = i / C;
  int r = (int)(t % R);
  int g = (int)(t / R);
  dst[(size_t)g * R * C + (size_t)c * R + r] = (bf16)src[i];
}

// ---------------------------------------------------------------- gates -----
__global__ void k_gate(const float* q_src, const float* k_src,
                       const float* sel_v, const float* sel_o,
                       float* w_rv, float* w_sv, float* w_ro, float* w_so,
                       int* list_v, int* list_o, int* cnt_v, int* cnt_o) {
  __shared__ float xs[DM];
  __shared__ float ps[256];
  __shared__ float gl[HE];
  const int m = blockIdx.x;
  const int g = blockIdx.y;           // 0: value gate (k_src,sel_v) 1: output gate (q_src,sel_o)
  const int t = threadIdx.x;
  const float* x = (g == 0) ? (k_src + (size_t)m * DM) : (q_src + (size_t)m * DM);
  const float* sel = (g == 0) ? sel_v : sel_o;
#pragma unroll
  for (int i = 0; i < 4; ++i) xs[t + 256*i] = x[t + 256*i];
  __syncthreads();
  const int he = t & 63, part = t >> 6;
  const float* srow = sel + (size_t)he * DM + part * 256;
  float acc = 0.f;
  for (int d = 0; d < 256; ++d) acc += xs[part*256 + d] * srow[d];
  ps[t] = acc;
  __syncthreads();
  if (t < HE) {
    float l = ps[t] + ps[t+64] + ps[t+128] + ps[t+192];
    gl[t] = 1.f / (1.f + expf(-l));
  }
  __syncthreads();
  if (t < H_) {
    const int h = t;
    float best = gl[h*E_]; int be = 0;
    for (int e = 1; e < RE; ++e) { float v = gl[h*E_+e]; if (v > best) { best = v; be = e; } }
    float* wr = (g == 0) ? w_rv : w_ro;
    float* ws = (g == 0) ? w_sv : w_so;
    int* list = (g == 0) ? list_v : list_o;
    int* cnt  = (g == 0) ? cnt_v  : cnt_o;
    wr[(size_t)m*H_ + h] = best;
    ws[(size_t)m*H_ + h] = gl[h*E_ + RE];
    int grp = h*RE + be;
    int slot = atomicAdd(&cnt[grp], 1);
    list[(size_t)grp * TOK + slot] = m;
  }
}

// ------------------------------------------------------- Q/K projection -----
__global__ void k_qkproj(const float* q_src, const float* k_src,
                         const float* Wq, const float* Wk,
                         bf16* qb, bf16* kb) {
  const int wave = threadIdx.x >> 5, lane = threadIdx.x & 31;
  const int ln = lane & 15, kh = lane >> 4, hi8 = kh * 8;
  const int m0 = blockIdx.x * 16;
  const int n0 = (blockIdx.y * 8 + wave) * 16;
  const float* src = blockIdx.z ? k_src : q_src;
  const float* W   = blockIdx.z ? Wk : Wq;
  bf16* dst        = blockIdx.z ? kb : qb;
  const float* arow = src + (size_t)(m0 + ln) * DM;
  const float* brow = W + (size_t)(n0 + ln) * DM;    // B[k,n] = W[n,k]
  v8f c = {};
  for (int kk = 0; kk < DM; kk += 32) {
    v16bf a = load_a_f32(arow, kk, kh);
    v16bf b;
    const float* bp = brow + kk + kh*16;
#pragma unroll
    for (int i = 0; i < 16; ++i) b[i] = (bf16)bp[i];
    c = wmma_bf16(a, b, c);
  }
  const float scale = 0.29730177875068026f;          // 128^-0.25
#pragma unroll
  for (int r = 0; r < 8; ++r) {
    int tok = m0 + r + hi8;
    int bi = tok >> 11, s = tok & (S_ - 1);
    int n = n0 + ln, h = n >> 7, f = n & 127;
    dst[(((size_t)(bi*H_ + h) * S_ + s) << 7) + f] = (bf16)(c[r] * scale);
  }
}

// ------------------------------------- MoE V projection: shared expert ------
__global__ void k_vshared(const float* v_src, const bf16* VT,
                          const float* w_sv, bf16* vT) {
  const int wave = threadIdx.x >> 5, lane = threadIdx.x & 31;
  const int ln = lane & 15, kh = lane >> 4, hi8 = kh * 8;
  const int m0 = blockIdx.x * 16;
  const int h  = blockIdx.y;
  const int n0 = wave * 16;                          // f tile
  const float* arow = v_src + (size_t)(m0 + ln) * DM;
  const bf16* brow = VT + ((size_t)(h*E_ + RE) * DH + (n0 + ln)) * (size_t)DM;
  v8f c = {};
  for (int kk = 0; kk < DM; kk += 32) {
    v16bf a = load_a_f32(arow, kk, kh);
    v16bf b = *(const v16bf*)(brow + kk + kh*16);
    c = wmma_bf16(a, b, c);
  }
  const int f = n0 + ln;
#pragma unroll
  for (int r = 0; r < 8; ++r) {
    int tok = m0 + r + hi8;
    int bi = tok >> 11, s = tok & (S_ - 1);
    float g = w_sv[(size_t)tok*H_ + h];
    vT[((size_t)(bi*H_ + h) * DH + f) * (size_t)S_ + s] = (bf16)(c[r] * g);
  }
}

// ------------------------------------- MoE V projection: routed expert ------
__global__ void k_vrouted(const float* v_src, const bf16* VT,
                          const float* w_rv, const int* list_v, const int* cnt_v,
                          bf16* vT) {
  const int grp = blockIdx.x >> 9;        // (h,e) group, 56 total
  const int tt  = blockIdx.x & 511;       // token tile within group
  const int cnt = cnt_v[grp];
  if (tt * 16 >= cnt) return;
  const int h = grp / RE, e = grp % RE;
  __shared__ int   stok[16];
  __shared__ float sw[16];
  if (threadIdx.x < 16) {
    int j = threadIdx.x, idx = tt*16 + j;
    if (idx < cnt) {
      int tok = list_v[(size_t)grp * TOK + idx];
      stok[j] = tok; sw[j] = w_rv[(size_t)tok*H_ + h];
    } else { stok[j] = -1; sw[j] = 0.f; }
  }
  __syncthreads();
  const int wave = threadIdx.x >> 5, lane = threadIdx.x & 31;
  const int ln = lane & 15, kh = lane >> 4, hi8 = kh * 8;
  const int n0 = wave * 16;
  int ta = stok[ln]; if (ta < 0) ta = 0;
  const float* arow = v_src + (size_t)ta * DM;
  const bf16* brow = VT + ((size_t)(h*E_ + e) * DH + (n0 + ln)) * (size_t)DM;
  v8f c = {};
  for (int kk = 0; kk < DM; kk += 32) {
    v16bf a = load_a_f32(arow, kk, kh);
    v16bf b = *(const v16bf*)(brow + kk + kh*16);
    c = wmma_bf16(a, b, c);
  }
  const int f = n0 + ln;
#pragma unroll
  for (int r = 0; r < 8; ++r) {
    int tok = stok[r + hi8];
    if (tok < 0) continue;
    int bi = tok >> 11, s = tok & (S_ - 1);
    size_t o = ((size_t)(bi*H_ + h) * DH + f) * (size_t)S_ + s;
    vT[o] = (bf16)((float)vT[o] + c[r] * sw[r + hi8]);   // unique rows -> no race
  }
}

// ------------------------------------------------- flash-style attention ----
// Block = 8 waves over 8 consecutive query tiles of the SAME (b,h):
// K/V chunks (32 keys) are staged once per block into LDS via async copies
// (double-buffered) and reused by all 8 waves.
__global__ void __launch_bounds__(256) k_attn(const bf16* qb, const bf16* kb,
                                              const bf16* vT, bf16* res) {
  __shared__ __align__(32) bf16 plds[8][16*32];      // per-wave P tile
  __shared__ __align__(32) bf16 kbuf[2][32*DH];      // [key][f]
  __shared__ __align__(32) bf16 vbuf[2][DH*32];      // [f][key]
  const int wave = threadIdx.x >> 5, lane = threadIdx.x & 31;
  const int ln = lane & 15, kh = lane >> 4, hi8 = kh * 8;
  const int bh = blockIdx.x >> 4;                    // 16 blocks per (b,h)
  const int qt = (blockIdx.x & 15) * 8 + wave;       // query tile in [0,128)
  const int m0 = qt * 16;
  const bf16* qbase = qb + (size_t)bh * S_ * DH;
  const bf16* kbase = kb + (size_t)bh * S_ * DH;
  const bf16* vbase = vT + (size_t)bh * DH * S_;
  const int t = threadIdx.x;

  v16bf qa[4];
  {
    const bf16* qrow = qbase + (size_t)(m0 + ln) * DH;
#pragma unroll
    for (int st = 0; st < 4; ++st) qa[st] = load_a_bf16(qrow, st*32, kh);
  }
  v8f acc[8];
  v8f z = {};
#pragma unroll
  for (int i = 0; i < 8; ++i) acc[i] = z;
  float mrun[8], lrun[8];
#pragma unroll
  for (int r = 0; r < 8; ++r) { mrun[r] = -3.0e38f; lrun[r] = 0.f; }
  bf16* pl = &plds[wave][0];

  // stage chunk 0 into buffer 0
  {
    const bf16* ksrc = kbase;                        // 32 keys * 128 f, contiguous 8KB
    stage16(&kbuf[0][t*8], ksrc + t*8);
    stage16(&kbuf[0][2048 + t*8], ksrc + 2048 + t*8);
#pragma unroll
    for (int q = 0; q < 2; ++q) {
      int id = t + q*256, f = id >> 2, part = id & 3;
      stage16(&vbuf[0][f*32 + part*8], vbase + (size_t)f * S_ + part*8);
    }
  }

  int p = 0;
  for (int kc = 0; kc < S_/32; ++kc, p ^= 1) {
    stage_wait();
    __syncthreads();                                 // fills visible; prev chunk fully consumed
    if (kc + 1 < S_/32) {                            // prefetch next chunk into other buffer
      const bf16* ksrc = kbase + (size_t)(kc + 1) * 32 * DH;
      stage16(&kbuf[p^1][t*8], ksrc + t*8);
      stage16(&kbuf[p^1][2048 + t*8], ksrc + 2048 + t*8);
#pragma unroll
      for (int q = 0; q < 2; ++q) {
        int id = t + q*256, f = id >> 2, part = id & 3;
        stage16(&vbuf[p^1][f*32 + part*8], vbase + (size_t)f * S_ + (kc + 1)*32 + part*8);
      }
      __builtin_prefetch(kbase + (size_t)(kc + 2) * 32 * DH + t*8, 0, 1);
    }
    // ---- scores: S = Q * K^T (16 x 32 chunk) ----
    v8f s0 = z, s1 = z;
#pragma unroll
    for (int st = 0; st < 4; ++st) {
      v16bf b0 = *(const v16bf*)(&kbuf[p][ln*DH + st*32 + kh*16]);
      v16bf b1 = *(const v16bf*)(&kbuf[p][(16 + ln)*DH + st*32 + kh*16]);
      s0 = wmma_bf16(qa[st], b0, s0);
      s1 = wmma_bf16(qa[st], b1, s1);
    }
    // ---- online softmax ----
    float resc[8];
#pragma unroll
    for (int r = 0; r < 8; ++r) {
      float mx = fmaxf(s0[r], s1[r]);
#pragma unroll
      for (int o = 8; o >= 1; o >>= 1) mx = fmaxf(mx, __shfl_xor(mx, o, 32));
      float mn = fmaxf(mrun[r], mx);
      resc[r] = expf(mrun[r] - mn);
      float p0 = expf(s0[r] - mn), p1 = expf(s1[r] - mn);
      float sum = p0 + p1;
#pragma unroll
      for (int o = 8; o >= 1; o >>= 1) sum += __shfl_xor(sum, o, 32);
      lrun[r] = lrun[r] * resc[r] + sum;
      mrun[r] = mn;
      pl[(r + hi8)*32 + ln]      = (bf16)p0;         // C-layout -> LDS (16x32 P tile)
      pl[(r + hi8)*32 + 16 + ln] = (bf16)p1;
    }
#pragma unroll
    for (int tt = 0; tt < 8; ++tt)
#pragma unroll
      for (int r = 0; r < 8; ++r) acc[tt][r] *= resc[r];
    __asm__ volatile("" ::: "memory");               // keep LDS store->load order (per-wave, in-order DS)
    v16bf pa = load_a_bf16(pl + ln*32, 0, kh);
    __asm__ volatile("" ::: "memory");
    // ---- acc += P * V ----
#pragma unroll
    for (int ft = 0; ft < 8; ++ft) {
      v16bf bv = *(const v16bf*)(&vbuf[p][(ft*16 + ln)*32 + kh*16]);
      acc[ft] = wmma_bf16(pa, bv, acc[ft]);
    }
  }
  const int b = bh >> 3, h = bh & 7;
#pragma unroll
  for (int r = 0; r < 8; ++r) {
    int srow = m0 + r + hi8;
    size_t tok = (size_t)b * S_ + srow;
    float inv = 1.f / lrun[r];
#pragma unroll
    for (int ft = 0; ft < 8; ++ft) {
      int f = ft*16 + ln;
      res[tok * DM + h*DH + f] = (bf16)(acc[ft][r] * inv);
    }
  }
}

// ------------------------------------- MoE O projection: shared expert ------
__global__ void k_oshared(const bf16* res, const bf16* OT, const float* w_so, float* out) {
  const int wave = threadIdx.x >> 5, lane = threadIdx.x & 31;
  const int ln = lane & 15, kh = lane >> 4, hi8 = kh * 8;
  const int m0 = blockIdx.x * 16;
  const int n0 = (blockIdx.y * 8 + wave) * 16;
  const bf16* arow = res + (size_t)(m0 + ln) * DM;
  const float* gs = w_so + (size_t)(m0 + ln) * H_;
  v8f c = {};
  for (int kk = 0; kk < DM; kk += 32) {
    const int h = kk >> 7;                  // head block of the contraction dim
    float g = gs[h];
    v8bf lo = *(const v8bf*)(arow + kk + kh*16);
    v8bf hi = *(const v8bf*)(arow + kk + kh*16 + 8);
    v16bf a;
#pragma unroll
    for (int i = 0; i < 8; ++i) {
      a[i]   = (bf16)((float)lo[i] * g);
      a[8+i] = (bf16)((float)hi[i] * g);
    }
    const bf16* bp = OT + ((size_t)(h*E_ + RE) * DM + (n0 + ln)) * (size_t)DH + (kk & 127) + kh*16;
    v16bf b = *(const v16bf*)bp;
    c = wmma_bf16(a, b, c);
  }
#pragma unroll
  for (int r = 0; r < 8; ++r)
    out[(size_t)(m0 + r + hi8) * DM + n0 + ln] = c[r];
}

// --------------------- MoE O projection: routed experts (one head/launch) ---
__global__ void k_orouted(const bf16* res, const bf16* OT,
                          const float* w_ro, const int* list_o, const int* cnt_o,
                          float* out, int h) {
  const int e  = blockIdx.x >> 9;
  const int tt = blockIdx.x & 511;
  const int grp = h*RE + e;
  const int cnt = cnt_o[grp];
  if (tt * 16 >= cnt) return;
  __shared__ int   stok[16];
  __shared__ float sw[16];
  if (threadIdx.x < 16) {
    int j = threadIdx.x, idx = tt*16 + j;
    if (idx < cnt) {
      int tok = list_o[(size_t)grp * TOK + idx];
      stok[j] = tok; sw[j] = w_ro[(size_t)tok*H_ + h];
    } else { stok[j] = -1; sw[j] = 0.f; }
  }
  __syncthreads();
  const int wave = threadIdx.x >> 5, lane = threadIdx.x & 31;
  const int ln = lane & 15, kh = lane >> 4, hi8 = kh * 8;
  const int n0 = (blockIdx.y * 8 + wave) * 16;
  int ta = stok[ln]; if (ta < 0) ta = 0;
  const bf16* arow = res + (size_t)ta * DM + h*DH;
  v8f c = {};
#pragma unroll
  for (int kk = 0; kk < DH; kk += 32) {
    v16bf a = load_a_bf16(arow, kk, kh);
    const bf16* bp = OT + ((size_t)(h*E_ + e) * DM + (n0 + ln)) * (size_t)DH + kk + kh*16;
    v16bf b = *(const v16bf*)bp;
    c = wmma_bf16(a, b, c);
  }
#pragma unroll
  for (int r = 0; r < 8; ++r) {
    int tok = stok[r + hi8];
    if (tok < 0) continue;
    size_t o = (size_t)tok * DM + n0 + ln;
    out[o] += c[r] * sw[r + hi8];           // rows unique within a head -> no race
  }
}

// ----------------------------------------------------------------------------
extern "C" void kernel_launch(void* const* d_in, const int* in_sizes, int n_in,
                              void* d_out, int out_size, void* d_ws, size_t ws_size,
                              hipStream_t stream) {
  (void)in_sizes; (void)n_in; (void)out_size; (void)ws_size;
  const float* q_src = (const float*)d_in[0];
  const float* k_src = (const float*)d_in[1];
  const float* v_src = (const float*)d_in[2];
  const float* Wq    = (const float*)d_in[3];
  const float* Wk    = (const float*)d_in[4];
  const float* Vw    = (const float*)d_in[5];
  const float* Ow    = (const float*)d_in[6];
  const float* sel_v = (const float*)d_in[7];
  const float* sel_o = (const float*)d_in[8];
  float* out = (float*)d_out;

  char* ws = (char*)d_ws;
  size_t off = 0;
  auto alloc = [&](size_t bytes) -> char* {
    char* p = ws + off;
    off = (off + bytes + 255) & ~(size_t)255;
    return p;
  };
  bf16* qb   = (bf16*)alloc((size_t)TOK * DM * 2);         // q bf16 [B,H,S,Dh]
  bf16* kb   = (bf16*)alloc((size_t)TOK * DM * 2);         // k bf16 [B,H,S,Dh]
  bf16* vT   = (bf16*)alloc((size_t)TOK * DM * 2);         // v bf16 [B,H,Dh,S] (transposed)
  bf16* resb = (bf16*)alloc((size_t)TOK * DM * 2);         // attn out bf16 [tok, H*Dh]
  bf16* VT   = (bf16*)alloc((size_t)HE * DM * DH * 2);     // V^T bf16 [he][f][k]
  bf16* OT   = (bf16*)alloc((size_t)HE * DM * DH * 2);     // O^T bf16 [he][d][f]
  float* w_rv = (float*)alloc((size_t)TOK * H_ * 4);
  float* w_sv = (float*)alloc((size_t)TOK * H_ * 4);
  float* w_ro = (float*)alloc((size_t)TOK * H_ * 4);
  float* w_so = (float*)alloc((size_t)TOK * H_ * 4);
  int* list_v = (int*)alloc((size_t)H_ * RE * TOK * 4);
  int* list_o = (int*)alloc((size_t)H_ * RE * TOK * 4);
  int* cnt_v  = (int*)alloc(256);
  int* cnt_o  = (int*)alloc(256);

  const size_t wtot = (size_t)HE * DM * DH;

  k_init<<<1, 64, 0, stream>>>(cnt_v, cnt_o);
  k_transpose<<<(unsigned)((wtot + 255) / 256), 256, 0, stream>>>(Vw, VT, DM, DH);
  k_transpose<<<(unsigned)((wtot + 255) / 256), 256, 0, stream>>>(Ow, OT, DH, DM);
  k_gate<<<dim3(TOK, 2), 256, 0, stream>>>(q_src, k_src, sel_v, sel_o,
                                           w_rv, w_sv, w_ro, w_so,
                                           list_v, list_o, cnt_v, cnt_o);
  k_qkproj<<<dim3(TOK/16, 8, 2), 256, 0, stream>>>(q_src, k_src, Wq, Wk, qb, kb);
  k_vshared<<<dim3(TOK/16, H_), 256, 0, stream>>>(v_src, VT, w_sv, vT);
  k_vrouted<<<dim3(H_*RE*512), 256, 0, stream>>>(v_src, VT, w_rv, list_v, cnt_v, vT);
  k_attn<<<dim3(B_*H_*16), 256, 0, stream>>>(qb, kb, vT, resb);
  k_oshared<<<dim3(TOK/16, 8), 256, 0, stream>>>(resb, OT, w_so, out);
  for (int h = 0; h < H_; ++h)
    k_orouted<<<dim3(RE*512, 8), 256, 0, stream>>>(resb, OT, w_ro, list_o, cnt_o, out, h);
}
